// RoIDelta_20057497272711
// MI455X (gfx1250) — compile-verified, matrix operand order checked by reading, not run on hardware
//
#include <hip/hip_runtime.h>

// RoI delta targets for MI455X (gfx1250, wave32).
// Roofline: output 27MB -> ~1.2us @ 23.3 TB/s. The only super-linear work is
// the exact top-64 ranking (32 batches x 2000^2 pairwise indicators = 128M),
// reduced on the matrix pipe via V_WMMA_I32_16X16X64_IU8 (indicator-matrix x
// all-ones = row counts). Stable argsort tie-breaking is preserved exactly by
// a composite u64 key (float bits || N-j): one v_cmp_gt_u64 per indicator.

#define BATCH   32
#define N_ROI   2000
#define N_GT    100
#define N_LAB   21
#define N_POS   64
#define I_TILES (N_ROI / 16)          // 125 (exact)
#define J_TILES ((N_ROI + 63) / 64)   // 32 (last tile partial: 16 valid)

typedef __attribute__((ext_vector_type(8))) int v8i;
typedef unsigned long long u64;

// ---------------------------------------------------------------------------
// Kernel 1: per-roi max IoU + argmax over gt boxes (gt staged in LDS).
// grid = (BATCH, ceil(N_ROI/256)), block = 256
// ---------------------------------------------------------------------------
__global__ void iou_argmax_kernel(const float* __restrict__ roi,
                                  const float* __restrict__ gt,
                                  float* __restrict__ merged,
                                  int* __restrict__ maxidx) {
    __shared__ float sg[N_GT * 4];
    __shared__ float sgArea[N_GT];
    const int b   = blockIdx.x;
    const int tid = threadIdx.x;

    for (int t = tid; t < N_GT * 4; t += blockDim.x)
        sg[t] = gt[(size_t)b * N_GT * 4 + t];
    __syncthreads();
    if (tid < N_GT) {
        float y1 = sg[tid*4+0], x1 = sg[tid*4+1];
        float y2 = sg[tid*4+2], x2 = sg[tid*4+3];
        sgArea[tid] = (y2 - y1) * (x2 - x1);
    }
    __syncthreads();

    const int i = blockIdx.y * blockDim.x + tid;
    if (i >= N_ROI) return;

    const float* rb = roi + ((size_t)b * N_ROI + i) * 4;
    const float by1 = rb[0], bx1 = rb[1], by2 = rb[2], bx2 = rb[3];
    const float bArea = (by2 - by1) * (bx2 - bx1);

    float best = -1.0f;   // IoU >= 0 always -> argmax picks first max (j asc)
    int bestj = 0;
    for (int j = 0; j < N_GT; ++j) {
        const float gy1 = sg[j*4+0], gx1 = sg[j*4+1];
        const float gy2 = sg[j*4+2], gx2 = sg[j*4+3];
        const float xt = fmaxf(bx1, gx1), yt = fmaxf(by1, gy1);
        const float xb = fminf(bx2, gx2), yb = fminf(by2, gy2);
        const float inter = fmaxf(xb - xt, 0.0f) * fmaxf(yb - yt, 0.0f);
        const float iou = inter / (bArea + sgArea[j] - inter);
        if (iou > best) { best = iou; bestj = j; }
    }
    merged[(size_t)b * N_ROI + i] = best;
    maxidx[(size_t)b * N_ROI + i] = bestj;
}

// ---------------------------------------------------------------------------
// Kernel 2: exact rank of each roi by descending IoU (stable ties: j < i).
//   rank[i] = #{ j : m[j] > m[i] } + #{ j : m[j]==m[i] && j<i }
// IoU >= 0, so float order == unsigned order of the bit pattern; composite key
//   key[j] = bits(m[j])<<32 | (N_ROI - j)   =>   ind(i,j) = key[j] > key[i]
// (one v_cmp_gt_u64 per indicator). Rank reduction on the matrix pipe:
//   A (16x64 IU8 indicators) x B (64x16 all-ones) += C (i32)  ->  D[:,n]=rank.
// One wave per 16-row i-tile; wave-uniform branches keep EXEC all-1s at WMMA.
// grid = (BATCH, ceil(I_TILES/8)), block = 256 (8 waves)
// ---------------------------------------------------------------------------
__global__ void rank_wmma_kernel(const float* __restrict__ merged,
                                 int* __restrict__ rank) {
    __shared__ u64 sk[N_ROI];                    // 16 KB of 320 KB/WGP
    const int b = blockIdx.x;
    for (int t = threadIdx.x; t < N_ROI; t += blockDim.x) {
        const unsigned mb = __float_as_uint(merged[(size_t)b * N_ROI + t]);
        sk[t] = ((u64)mb << 32) | (unsigned)(N_ROI - t);   // low word >= 1
    }
    __syncthreads();

    const int wave = threadIdx.x >> 5;
    const int lane = threadIdx.x & 31;
    const int tile = blockIdx.y * 8 + wave;      // uniform within a wave
    if (tile >= I_TILES) return;                 // wave-uniform exit

    const int M     = lane & 15;                 // A-matrix row for this lane
    const int half  = lane >> 4;                 // K-half per ISA A-layout
    const int iGlob = tile * 16 + M;
    const u64 ki    = sk[iGlob];

    v8i c = {};                                  // i32 accumulator (zero)
    v8i bOnes;
#pragma unroll
    for (int v = 0; v < 8; ++v) bOnes[v] = 0x01010101;   // B == 1 everywhere

    // 31 full tiles: no bounds check in the hot path.
    for (int jt = 0; jt < J_TILES - 1; ++jt) {
        v8i a;
#pragma unroll
        for (int v = 0; v < 8; ++v) {
            // ISA 8-bit A 16x64 layout: K = (v/2)*16 + half*8 + (v&1)*4 + t
            const int j0 = jt * 64 + ((v >> 1) << 4) + (half << 3) + ((v & 1) << 2);
            unsigned pack = 0;
#pragma unroll
            for (int t = 0; t < 4; ++t)
                pack |= (sk[j0 + t] > ki ? 1u : 0u) << (8 * t);
            a[v] = (int)pack;
        }
        c = __builtin_amdgcn_wmma_i32_16x16x64_iu8(
                false, a, false, bOnes, c, false, false);
    }
    // Tail tile (j = 1984..2047; only j < 2000 valid -> key 0 < ki always).
    {
        v8i a;
#pragma unroll
        for (int v = 0; v < 8; ++v) {
            const int j0 = (J_TILES - 1) * 64
                         + ((v >> 1) << 4) + (half << 3) + ((v & 1) << 2);
            unsigned pack = 0;
#pragma unroll
            for (int t = 0; t < 4; ++t) {
                const int j = j0 + t;
                const u64 kj = (j < N_ROI) ? sk[j] : 0ull;
                pack |= (kj > ki ? 1u : 0u) << (8 * t);
            }
            a[v] = (int)pack;
        }
        c = __builtin_amdgcn_wmma_i32_16x16x64_iu8(
                false, a, false, bOnes, c, false, false);
    }

    // C/D i32 16x16 layout: lane 0 (N=0) VGPR r -> row r; lane 16 -> row 8+r
    if (lane == 0) {
#pragma unroll
        for (int r = 0; r < 8; ++r)
            rank[(size_t)b * N_ROI + tile * 16 + r] = c[r];
    } else if (lane == 16) {
#pragma unroll
        for (int r = 0; r < 8; ++r)
            rank[(size_t)b * N_ROI + tile * 16 + 8 + r] = c[r];
    }
}

// ---------------------------------------------------------------------------
// Kernel 3: one thread per (b, i, c): deltas row (float4, coalesced) + label.
// Non-pos rois: gt box = 0 -> deltas 0, label one-hot class 0 (matches ref).
// grid = ceil(BATCH*N_ROI*N_LAB/256), block = 256
// ---------------------------------------------------------------------------
__global__ void finalize_kernel(const float* __restrict__ roi,
                                const float* __restrict__ gt,
                                const int* __restrict__ gtl,
                                const int* __restrict__ rank,
                                const int* __restrict__ maxidx,
                                float* __restrict__ out) {
    const int total = BATCH * N_ROI * N_LAB;
    const int gid = blockIdx.x * blockDim.x + threadIdx.x;
    if (gid >= total) return;

    const int c  = gid % N_LAB;
    const int bi = gid / N_LAB;          // b*N_ROI + i
    const int b  = bi / N_ROI;

    const bool pos = rank[bi] < N_POS;
    const int  mj  = maxidx[bi];
    const int  lab = pos ? gtl[b * N_GT + mj] : 0;

    const float* rb = roi + (size_t)bi * 4;
    const float ry1 = rb[0], rx1 = rb[1], ry2 = rb[2], rx2 = rb[3];

    float gy1 = 0.f, gx1 = 0.f, gy2 = 0.f, gx2 = 0.f;
    if (pos) {
        const float* gb = gt + ((size_t)b * N_GT + mj) * 4;
        gy1 = gb[0]; gx1 = gb[1]; gy2 = gb[2]; gx2 = gb[3];
    }

    float bw = rx2 - rx1, bh = ry2 - ry1;
    const float bcx = rx1 + 0.5f * bw, bcy = ry1 + 0.5f * bh;
    const float gw = gx2 - gx1, gh = gy2 - gy1;
    const float gcx = gx1 + 0.5f * gw, gcy = gy1 + 0.5f * gh;
    bw = (bw == 0.0f) ? 1e-3f : bw;
    bh = (bh == 0.0f) ? 1e-3f : bh;
    const float dx = (gw == 0.0f) ? 0.0f : (gcx - bcx) / bw;
    const float dy = (gh == 0.0f) ? 0.0f : (gcy - bcy) / bh;
    const float dw = (gw == 0.0f) ? 0.0f : logf(gw / bw);
    const float dh = (gh == 0.0f) ? 0.0f : logf(gh / bh);

    const float oneh = (c == lab) ? 1.0f : 0.0f;
    float4 dv;
    dv.x = oneh * dy; dv.y = oneh * dx; dv.z = oneh * dh; dv.w = oneh * dw;
    ((float4*)out)[gid] = dv;                       // deltas: [total*4] floats
    out[(size_t)total * 4 + gid] = oneh;            // labels: [total] floats
}

// ---------------------------------------------------------------------------
extern "C" void kernel_launch(void* const* d_in, const int* in_sizes, int n_in,
                              void* d_out, int out_size, void* d_ws, size_t ws_size,
                              hipStream_t stream) {
    (void)in_sizes; (void)n_in; (void)out_size; (void)ws_size;
    const float* roi = (const float*)d_in[0];   // (32, 2000, 4) f32
    const float* gt  = (const float*)d_in[1];   // (32, 100, 4)  f32
    const int*   gtl = (const int*)d_in[2];     // (32, 100)     i32
    float* out = (float*)d_out;

    // workspace: merged f32 [32*2000] | maxidx i32 [32*2000] | rank i32 [32*2000]
    float* merged = (float*)d_ws;
    int*   maxidx = (int*)(merged + BATCH * N_ROI);
    int*   rnk    = maxidx + BATCH * N_ROI;

    dim3 blk(256);
    dim3 g1(BATCH, (N_ROI + 255) / 256);
    iou_argmax_kernel<<<g1, blk, 0, stream>>>(roi, gt, merged, maxidx);

    dim3 g2(BATCH, (I_TILES + 7) / 8);
    rank_wmma_kernel<<<g2, blk, 0, stream>>>(merged, rnk);

    const int total = BATCH * N_ROI * N_LAB;    // 1,344,000
    dim3 g3((total + 255) / 256);
    finalize_kernel<<<g3, blk, 0, stream>>>(roi, gt, gtl, rnk, maxidx, out);
}
